// MLPAttention_80247168958959
// MI455X (gfx1250) — matrix operand order repaired and checked by reading, not compiled, not run on hardware
//
#include <hip/hip_runtime.h>
#include <hip/hip_bf16.h>
#include <math.h>

// Problem dims (fixed by reference)
#define BQ 8
#define QL 128
#define KL 512
#define DD 256
#define UU 256
#define NEG_INF_VAL (-1e6f)

typedef __attribute__((ext_vector_type(16))) __bf16 v16bf;
typedef __attribute__((ext_vector_type(8)))  float  v8f;

// ---------------- WMMA fragment helpers (wave32, CDNA5 layouts) ----------------
// A-matrix 16x32 bf16 (M x K), per ISA 7.12.2:
//   lanes 0-15: M=lane, K-base 0 ; lanes 16-31: M=lane-16, K-base 8
//   elements j=0..7 -> K = kb + j ; j=8..15 -> K = 16 + kb + (j-8)
__device__ inline v16bf load_a_frag_f32(const float* __restrict__ src,
                                        int row0, int k0, int ld) {
  int lane = threadIdx.x & 31;
  int m  = lane & 15;
  int kb = (lane >> 4) * 8;
  const float* p = src + (size_t)(row0 + m) * ld + k0 + kb;
  v16bf a;
#pragma unroll
  for (int j = 0; j < 16; ++j) {
    int K = j + ((j >= 8) ? 8 : 0);
    a[j] = (__bf16)p[K];
  }
  return a;
}

// B-matrix 32x16 bf16 (K x N): one row (fixed K) striped across lanes per VGPR.
//   lanes 0-15: N=lane, K-base 0 ; lanes 16-31: N=lane-16, K-base 8; j<8 -> K=kb+j, j>=8 -> K=16+kb+(j-8)
__device__ inline v16bf load_b_frag_f32(const float* __restrict__ src,
                                        int k0, int n0, int ld) {
  int lane = threadIdx.x & 31;
  int n    = lane & 15;
  int koff = (lane >> 4) * 8;
  const float* p = src + (size_t)(k0 + koff) * ld + n0 + n;
  v16bf b;
#pragma unroll
  for (int j = 0; j < 16; ++j) b[j] = (__bf16)p[(size_t)(j + ((j >= 8) ? 8 : 0)) * ld];
  return b;
}

// C/D 16x16 f32: VGPR r -> M = r (lanes 0-15) or r+8 (lanes 16-31), N = lane%16
__device__ inline void store_cd(float* __restrict__ dst, int row0, int n0,
                                int ld, v8f c) {
  int lane = threadIdx.x & 31;
  int n  = lane & 15;
  int mb = (lane >> 4) * 8;
#pragma unroll
  for (int r = 0; r < 8; ++r)
    dst[(size_t)(row0 + mb + r) * ld + n0 + n] = c[r];
}

// ---------------- Kernel 1: fused q/k projection (bf16 WMMA GEMM) ----------------
// Combined A: rows [0,1024)=query(Wq->qp), rows [1024,5120)=key(Wk->kp).
// Wave computes a 16x64 strip (4 accumulators, A-frag reused 4x).
// Jobs = 320 row-tiles * 4 col-groups = 1280 waves -> 160 blocks.
__global__ void __launch_bounds__(256)
proj_kernel(const float* __restrict__ query, const float* __restrict__ key_,
            const float* __restrict__ Wq,    const float* __restrict__ Wk,
            float* __restrict__ qp, float* __restrict__ kp) {
  int wave = threadIdx.x >> 5;
  int job  = blockIdx.x * 8 + wave;     // 0..1279
  int r0   = (job >> 2) * 16;           // global row tile
  int n0   = (job & 3) * 64;            // U column group (4 x 16)
  const float *A, *W;
  float* D;
  int arow;
  if (r0 < BQ * QL) { A = query; W = Wq; D = qp; arow = r0; }
  else              { A = key_;  W = Wk; D = kp; arow = r0 - BQ * QL; }
  v8f acc[4] = {};
#pragma unroll
  for (int kk = 0; kk < DD; kk += 32) {
    v16bf a = load_a_frag_f32(A, arow, kk, DD);
#pragma unroll
    for (int t = 0; t < 4; ++t) {
      v16bf b = load_b_frag_f32(W, kk, n0 + 16 * t, UU);
      acc[t] = __builtin_amdgcn_wmma_f32_16x16x32_bf16(false, a, false, b,
                                                       (short)0, acc[t],
                                                       false, false);
    }
  }
#pragma unroll
  for (int t = 0; t < 4; ++t) store_cd(D, arow, n0 + 16 * t, UU, acc[t]);
}

// ---------------- Kernel 2: tanh core (VALU/TRANS + async-to-LDS staging) -------
// scores[b,q,k] = sum_u v_w[u] * tanh(qp[b,q,u] + kp[b,k,u]); mask k>=valid_len.
// Block = (kchunk, q, b); 256 threads each own one k. kp tile (256 k x 64 u)
// staged with GLOBAL_LOAD_ASYNC_TO_LDS_B128 (ASYNCcnt), rows padded to 65
// floats -> conflict-free on 64 banks for both 16B stores and row reads.
__global__ void __launch_bounds__(256)
score_kernel(const float* __restrict__ qp, const float* __restrict__ kp,
             const float* __restrict__ v_w, const int* __restrict__ valid_len,
             float* __restrict__ scores) {
  __shared__ float qrow[UU];
  __shared__ float vw[UU];
  __shared__ float kpt[256 * 65];

  int tid    = threadIdx.x;
  int kchunk = blockIdx.x;   // 0..1
  int q      = blockIdx.y;   // 0..127
  int b      = blockIdx.z;   // 0..7
  int k0     = kchunk * 256;

  qrow[tid] = qp[((size_t)b * QL + q) * UU + tid];
  vw[tid]   = v_w[tid];

  // staging coords: 16 rows x 16 lanes-of-16B per iteration group
  int rk = tid >> 4;          // 0..15: row within 16-row group
  int cu = (tid & 15) * 4;    // float column, multiple of 4

  const float* gchunk = kp + ((size_t)b * KL + k0) * UU;  // [256 x 256] tile base
  unsigned lds0 = (unsigned)(uintptr_t)&kpt[0];

  float acc = 0.0f;
  for (int c = 0; c < 4; ++c) {
    __syncthreads();  // previous chunk fully consumed before overwrite
    const float* gbase = gchunk + c * 64;
#pragma unroll
    for (int it = 0; it < 16; ++it) {
      int kk       = it * 16 + rk;
      int goff     = kk * (int)(UU * sizeof(float)) + cu * (int)sizeof(float);
      unsigned lds = lds0 + (unsigned)((kk * 65 + cu) * sizeof(float));
      asm volatile("global_load_async_to_lds_b128 %0, %1, %2 offset:0"
                   :: "v"(lds), "v"(goff), "s"(gbase)
                   : "memory");
    }
    asm volatile("s_wait_asynccnt 0x0" ::: "memory");
    __syncthreads();

    if (c < 3)  // warm L2/GL1 for the next u-chunk (global_prefetch_b8)
      __builtin_prefetch(gchunk + (c + 1) * 64 + (size_t)tid * UU, 0, 1);

    const float* krow = &kpt[tid * 65];
#pragma unroll 8
    for (int uu = 0; uu < 64; ++uu) {
      float t = tanhf(qrow[c * 64 + uu] + krow[uu]);
      acc = fmaf(vw[c * 64 + uu], t, acc);
    }
  }
  int k = k0 + tid;
  float s = (k < valid_len[b]) ? acc : NEG_INF_VAL;
  scores[((size_t)b * QL + q) * KL + k] = s;
}

// ---------------- Kernel 3: softmax over K (in-place) ----------------
__global__ void __launch_bounds__(256)
softmax_kernel(float* __restrict__ scores) {
  __shared__ float red[256];
  float* s = scores + (size_t)blockIdx.x * KL;   // one (b,q) row
  int tid = threadIdx.x;

  float x0 = s[tid], x1 = s[tid + 256];
  red[tid] = fmaxf(x0, x1);
  __syncthreads();
  for (int off = 128; off > 0; off >>= 1) {
    if (tid < off) red[tid] = fmaxf(red[tid], red[tid + off]);
    __syncthreads();
  }
  float m = red[0];
  __syncthreads();

  float e0 = __expf(x0 - m), e1 = __expf(x1 - m);
  red[tid] = e0 + e1;
  __syncthreads();
  for (int off = 128; off > 0; off >>= 1) {
    if (tid < off) red[tid] += red[tid + off];
    __syncthreads();
  }
  float inv = 1.0f / red[0];
  s[tid]       = e0 * inv;
  s[tid + 256] = e1 * inv;
}

// ---------------- Kernel 4: out = attn @ value (bf16 WMMA GEMM) ----------------
// Per batch: [128 x 512] @ [512 x 256]. Wave computes 16x64 strip (4 acc).
// Jobs = 8b * 8qt * 4dg = 256 waves -> 32 blocks.
__global__ void __launch_bounds__(256)
av_kernel(const float* __restrict__ attn, const float* __restrict__ value,
          float* __restrict__ out) {
  int wave = threadIdx.x >> 5;
  int job  = blockIdx.x * 8 + wave;     // 0..255
  int b  = job >> 5;
  int q0 = ((job >> 2) & 7) * 16;
  int d0 = (job & 3) * 64;

  const float* A = attn  + ((size_t)b * QL + q0) * KL;
  const float* V = value + (size_t)b * KL * DD;
  v8f acc[4] = {};
#pragma unroll 4
  for (int kk = 0; kk < KL; kk += 32) {
    v16bf a = load_a_frag_f32(A, 0, kk, KL);
#pragma unroll
    for (int t = 0; t < 4; ++t) {
      v16bf bb = load_b_frag_f32(V, kk, d0 + 16 * t, DD);
      acc[t] = __builtin_amdgcn_wmma_f32_16x16x32_bf16(false, a, false, bb,
                                                       (short)0, acc[t],
                                                       false, false);
    }
  }
#pragma unroll
  for (int t = 0; t < 4; ++t)
    store_cd(out + (size_t)b * QL * DD, q0, d0 + 16 * t, DD, acc[t]);
}

// ---------------- host launch ----------------
extern "C" void kernel_launch(void* const* d_in, const int* in_sizes, int n_in,
                              void* d_out, int out_size, void* d_ws, size_t ws_size,
                              hipStream_t stream) {
  const float* query     = (const float*)d_in[0];
  const float* key_      = (const float*)d_in[1];
  const float* value     = (const float*)d_in[2];
  const int*   valid_len = (const int*)  d_in[3];
  const float* Wq        = (const float*)d_in[4];
  const float* Wk        = (const float*)d_in[5];
  const float* v_w       = (const float*)d_in[6];
  float* out = (float*)d_out;

  // workspace layout (floats): qp[1024*256] | kp[4096*256] | scores[8*128*512] ~ 7MB
  float* ws     = (float*)d_ws;
  float* qp     = ws;
  float* kp     = qp + (size_t)BQ * QL * UU;
  float* scores = kp + (size_t)BQ * KL * UU;

  proj_kernel<<<160, 256, 0, stream>>>(query, key_, Wq, Wk, qp, kp);

  dim3 sgrid(KL / 256, QL, BQ);   // (2, 128, 8) = 2048 blocks
  score_kernel<<<sgrid, 256, 0, stream>>>(qp, kp, v_w, valid_len, scores);

  softmax_kernel<<<BQ * QL, 256, 0, stream>>>(scores);

  av_kernel<<<32, 256, 0, stream>>>(scores, value, out);
}
